// attention_unit_61589831024829
// MI455X (gfx1250) — matrix operand order repaired and checked by reading, not compiled, not run on hardware
//
// DIN attention unit for MI455X (gfx1250): f16 WMMA GEMMs + fused dice/softmax pipeline.
// GEMM B-operands fetched directly from pre-transposed f16 weights (L2-resident),
// A-operands staged per-wave in LDS with exact ISA 16x32 f16 layout.
#include <hip/hip_runtime.h>

typedef __attribute__((ext_vector_type(16))) _Float16 v16h;
typedef __attribute__((ext_vector_type(8)))  _Float16 v8h;
typedef __attribute__((ext_vector_type(4)))  _Float16 v4h;
typedef __attribute__((ext_vector_type(8)))  float    v8f;
typedef __attribute__((ext_vector_type(4)))  float    v4f;

constexpr int kS    = 200;          // seq len (reduction axis for dice/softmax)
constexpr int kB    = 1024;         // batch
constexpr int kD    = 128;          // feature dim of facts/query
constexpr int kF    = 512;          // GEMM1 K
constexpr int kL1   = 80;
constexpr int kL2   = 40;
constexpr int kRows = kS * kB;      // 204800 flattened (s,b) rows
constexpr float kMaskFill = -4294967295.0f;   // rounds to -2^32 in f32, same as jnp

__device__ __forceinline__ float sigmoidf_(float z) {
  return 1.0f / (1.0f + __expf(-z));
}

__device__ __forceinline__ v16h cat8(v8h lo, v8h hi) {
  return __builtin_shufflevector(lo, hi, 0,1,2,3,4,5,6,7,8,9,10,11,12,13,14,15);
}

__device__ __forceinline__ v8f wmma_f16(v16h a, v16h b, v8f c) {
  return __builtin_amdgcn_wmma_f32_16x16x32_f16(false, a, false, b, (short)0, c,
                                                false, false);
}

// ---------------------------------------------------------------------------
// Prep: W1 (512x80 f32) -> w1t[80][512] f16 ; W2 (80x40) -> w2t[48][96] f16,
// zero-padded so GEMM2 can run K=96 / N=48 tiles. Both are tiny & L2-resident.
// ---------------------------------------------------------------------------
__global__ __launch_bounds__(256) void attn_prep_weights(
    const float* __restrict__ W1, const float* __restrict__ W2,
    _Float16* __restrict__ w1t, _Float16* __restrict__ w2t)
{
  const int idx = blockIdx.x * blockDim.x + threadIdx.x;
  if (idx < kL1 * kF) {
    const int n = idx / kF, k = idx % kF;
    w1t[idx] = (_Float16)W1[k * kL1 + n];
  }
  const int j = idx - kL1 * kF;
  if (j >= 0 && j < 48 * 96) {
    const int n = j / 96, k = j % 96;
    w2t[j] = (_Float16)((k < kL1 && n < kL2) ? W2[k * kL2 + n] : 0.f);
  }
}

// ---------------------------------------------------------------------------
// Kernel 1: h1[r, 0:80] = x[r, 0:512] @ W1 + b1   (f32 accum via v_wmma f16)
// x[r,k] = { query[r/S, k], facts[r, k-128], q*f, q-f } per 128-col segment
// (torch-faithful tiled-query row index (s*B+b)/S = r/S).
// One wave per block; each wave owns a 32x80 tile (two 16-row A tiles so every
// B fetch feeds two WMMAs). B comes straight from w1t in global (L2 hit).
// ---------------------------------------------------------------------------
__global__ __launch_bounds__(32) void attn_gemm1(
    const float* __restrict__ query, const float* __restrict__ facts,
    const _Float16* __restrict__ w1t, const float* __restrict__ b1,
    _Float16* __restrict__ h1)
{
  __shared__ alignas(16) _Float16 xlds[32][520];  // pitch 1040B: 16B-aligned, conflict-free

  const int lane = threadIdx.x;
  const int r0   = blockIdx.x * 32;

  // Stage x tile as f16: float4 loads, packed b64 LDS stores.
  for (int idx = lane; idx < 32 * 32; idx += 32) {
    const int m = idx >> 5;
    const int c = (idx & 31) << 2;            // column (multiple of 4)
    const int r = r0 + m;
    const int qi = (int)((unsigned)r / (unsigned)kS);
    const v4f f = *(const v4f*)(facts + (size_t)r * kD + c);
    const v4f q = *(const v4f*)(query + (size_t)qi * kD + c);
    v4h hq, hf, hp, hd;
#pragma unroll
    for (int j = 0; j < 4; ++j) {
      hq[j] = (_Float16)q[j];
      hf[j] = (_Float16)f[j];
      hp[j] = (_Float16)(q[j] * f[j]);
      hd[j] = (_Float16)(q[j] - f[j]);
    }
    *(v4h*)&xlds[m][      c] = hq;
    *(v4h*)&xlds[m][128 + c] = hf;
    *(v4h*)&xlds[m][256 + c] = hp;
    *(v4h*)&xlds[m][384 + c] = hd;
  }
  __syncthreads();   // single-wave fence: order LDS stores before cross-lane reads

  v8f acc[2][5];
#pragma unroll
  for (int u = 0; u < 2; ++u)
#pragma unroll
    for (int t = 0; t < 5; ++t)
      acc[u][t] = v8f{0.f,0.f,0.f,0.f,0.f,0.f,0.f,0.f};

  const int mrow  = lane & 15;
  const int khalf = (lane >> 4) << 3;          // A: 0 (lanes 0-15) / 8 (lanes 16-31)
  const int k0b   = (lane < 16) ? 0 : 16;      // unused; B takes its K base from khalf

  for (int ks = 0; ks < 16; ++ks) {
    const int kb = ks * 32;
    // A operands (ISA 16x32 f16 layout): two ds_load_b128 each.
    const _Float16* x0 = &xlds[mrow][kb];
    const _Float16* x1 = &xlds[mrow + 16][kb];
    const v16h A0 = cat8(*(const v8h*)(x0 + khalf), *(const v8h*)(x0 + 16 + khalf));
    const v16h A1 = cat8(*(const v8h*)(x1 + khalf), *(const v8h*)(x1 + 16 + khalf));
#pragma unroll
    for (int nt = 0; nt < 5; ++nt) {
      // B operand (32x16): K khalf..khalf+7 and 16+khalf..16+khalf+7 at N = nt*16 + lane%16, from global f16.
      const _Float16* wrow = w1t + (size_t)(nt * 16 + mrow) * kF + kb + khalf;
      const v16h Bv = cat8(*(const v8h*)wrow, *(const v8h*)(wrow + 16));
      acc[0][nt] = wmma_f16(A0, Bv, acc[0][nt]);
      acc[1][nt] = wmma_f16(A1, Bv, acc[1][nt]);
    }
  }

  // C/D layout: VGPR i -> M = i (+8 for lanes>=16), N = lane%16.
#pragma unroll
  for (int nt = 0; nt < 5; ++nt) {
    const int n = nt * 16 + mrow;
    const float bias = b1[n];
#pragma unroll
    for (int i = 0; i < 8; ++i) {
      const int m0 = i + ((lane >> 4) << 3);
      h1[(size_t)(r0 + m0     ) * kL1 + n] = (_Float16)(acc[0][nt][i] + bias);
      h1[(size_t)(r0 + 16 + m0) * kL1 + n] = (_Float16)(acc[1][nt][i] + bias);
    }
  }
}

// ---------------------------------------------------------------------------
// Mean / reciprocal unbiased std over S per (b, l). Deterministic fixed-order.
// ---------------------------------------------------------------------------
__global__ __launch_bounds__(256) void attn_stats(
    const _Float16* __restrict__ h, float* __restrict__ mean,
    float* __restrict__ rstd, int L)
{
  const int idx = blockIdx.x * blockDim.x + threadIdx.x;   // b*L + l
  if (idx >= kB * L) return;
  const size_t stride = (size_t)kB * L;
  float s = 0.f, s2 = 0.f;
  size_t p = (size_t)idx;
  for (int t = 0; t < kS; ++t) {
    const float v = (float)h[p];
    s += v; s2 += v * v;
    p += stride;
  }
  const float m = s / (float)kS;
  const float var = (s2 - s * m) / (float)(kS - 1);
  mean[idx] = m;
  rstd[idx] = rsqrtf(var);
}

// ---------------------------------------------------------------------------
// Kernel 3: h2 = dice(h1) @ W2 + b2. K padded 80->96, N padded 40->48 (zeros
// provided by w2t padding). Same 32-row/wave scheme as GEMM1.
// ---------------------------------------------------------------------------
__global__ __launch_bounds__(32) void attn_gemm2(
    const _Float16* __restrict__ h1, const float* __restrict__ mean1,
    const float* __restrict__ rstd1, const float* __restrict__ a1,
    const _Float16* __restrict__ w2t, const float* __restrict__ b2,
    _Float16* __restrict__ h2)
{
  __shared__ alignas(16) _Float16 alds[32][104];   // pitch 208B, conflict-free

  const int lane = threadIdx.x;
  const int r0   = blockIdx.x * 32;
  const float alpha = a1[0];

  // dice(h1) tile, zero-padded K: v4h loads of h1, packed b64 LDS stores.
  for (int idx = lane; idx < 32 * 24; idx += 32) {
    const int m = idx / 24;
    const int c = (idx % 24) << 2;               // 0..92 step 4
    const int r = r0 + m;
    v4h v = {};
    if (c < kL1) {
      const int b  = r & (kB - 1);
      const int si = b * kL1 + c;
      const v4h xh = *(const v4h*)(h1 + (size_t)r * kL1 + c);
#pragma unroll
      for (int j = 0; j < 4; ++j) {
        const float x = (float)xh[j];
        const float p = sigmoidf_((x - mean1[si + j]) * rstd1[si + j]);
        v[j] = (_Float16)(x * p + alpha * x * (1.f - p));
      }
    }
    *(v4h*)&alds[m][c] = v;
  }
  __syncthreads();

  v8f acc[2][3];
#pragma unroll
  for (int u = 0; u < 2; ++u)
#pragma unroll
    for (int t = 0; t < 3; ++t)
      acc[u][t] = v8f{0.f,0.f,0.f,0.f,0.f,0.f,0.f,0.f};

  const int mrow  = lane & 15;
  const int khalf = (lane >> 4) << 3;
  const int k0b   = (lane < 16) ? 0 : 16;

#pragma unroll
  for (int ks = 0; ks < 3; ++ks) {
    const int kb = ks * 32;
    const _Float16* x0 = &alds[mrow][kb];
    const _Float16* x1 = &alds[mrow + 16][kb];
    const v16h A0 = cat8(*(const v8h*)(x0 + khalf), *(const v8h*)(x0 + 16 + khalf));
    const v16h A1 = cat8(*(const v8h*)(x1 + khalf), *(const v8h*)(x1 + 16 + khalf));
#pragma unroll
    for (int nt = 0; nt < 3; ++nt) {
      const _Float16* wrow = w2t + (size_t)(nt * 16 + mrow) * 96 + kb + khalf;
      const v16h Bv = cat8(*(const v8h*)wrow, *(const v8h*)(wrow + 16));
      acc[0][nt] = wmma_f16(A0, Bv, acc[0][nt]);
      acc[1][nt] = wmma_f16(A1, Bv, acc[1][nt]);
    }
  }

#pragma unroll
  for (int nt = 0; nt < 3; ++nt) {
    const int n = nt * 16 + mrow;
    if (n < kL2) {
      const float bias = b2[n];
#pragma unroll
      for (int i = 0; i < 8; ++i) {
        const int m0 = i + ((lane >> 4) << 3);
        h2[(size_t)(r0 + m0     ) * kL2 + n] = (_Float16)(acc[0][nt][i] + bias);
        h2[(size_t)(r0 + 16 + m0) * kL2 + n] = (_Float16)(acc[1][nt][i] + bias);
      }
    }
  }
}

// ---------------------------------------------------------------------------
// Kernel 5: logits[r] = mask ? dice(h2[r]) . W3 + b3 : MASK_FILL
// ---------------------------------------------------------------------------
__global__ __launch_bounds__(256) void attn_logits(
    const _Float16* __restrict__ h2, const float* __restrict__ mean2,
    const float* __restrict__ rstd2, const float* __restrict__ a2,
    const float* __restrict__ W3, const float* __restrict__ b3,
    const unsigned char* __restrict__ mask, float* __restrict__ logits)
{
  const int r = blockIdx.x * blockDim.x + threadIdx.x;
  if (r >= kRows) return;
  const int b = r & (kB - 1);
  const float alpha = a2[0];
  float acc = 0.f;
#pragma unroll
  for (int c = 0; c < kL2; c += 4) {
    const v4h xh = *(const v4h*)(h2 + (size_t)r * kL2 + c);
    const int si = b * kL2 + c;
#pragma unroll
    for (int j = 0; j < 4; ++j) {
      const float x = (float)xh[j];
      const float p = sigmoidf_((x - mean2[si + j]) * rstd2[si + j]);
      acc += (x * p + alpha * x * (1.f - p)) * W3[c + j];
    }
  }
  acc += b3[0];
  logits[r] = mask[r] ? acc : kMaskFill;
}

// ---------------------------------------------------------------------------
// Kernel 6: softmax over S per batch column + scale facts. One WG per b.
// ---------------------------------------------------------------------------
__global__ __launch_bounds__(256) void attn_softmax_scale(
    const float* __restrict__ logits, const float* __restrict__ facts,
    float* __restrict__ out)
{
  __shared__ float red[256];
  __shared__ float wts[256];
  const int b = blockIdx.x;
  const int tid = threadIdx.x;

  float mx = -3.402823466e38f;
  for (int s = tid; s < kS; s += 256) mx = fmaxf(mx, logits[s * kB + b]);
  red[tid] = mx; __syncthreads();
  for (int off = 128; off > 0; off >>= 1) {
    if (tid < off) red[tid] = fmaxf(red[tid], red[tid + off]);
    __syncthreads();
  }
  mx = red[0]; __syncthreads();

  float sm = 0.f;
  for (int s = tid; s < kS; s += 256) sm += __expf(logits[s * kB + b] - mx);
  red[tid] = sm; __syncthreads();
  for (int off = 128; off > 0; off >>= 1) {
    if (tid < off) red[tid] += red[tid + off];
    __syncthreads();
  }
  const float inv = 1.f / red[0];
  if (tid < kS) wts[tid] = __expf(logits[tid * kB + b] - mx) * inv;
  __syncthreads();

  // float4 streaming: out[s,b,:] = w(s,b) * facts[s,b,:]
  for (int idx = tid; idx < kS * (kD / 4); idx += 256) {
    const int s  = idx >> 5;                 // /32 chunks per row
    const int d4 = (idx & 31) << 2;
    const size_t o = ((size_t)s * kB + b) * kD + d4;
    const v4f f = *(const v4f*)(facts + o);
    const float w = wts[s];
    v4f res;
#pragma unroll
    for (int j = 0; j < 4; ++j) res[j] = w * f[j];
    *(v4f*)(out + o) = res;
  }
}

// ---------------------------------------------------------------------------
extern "C" void kernel_launch(void* const* d_in, const int* in_sizes, int n_in,
                              void* d_out, int out_size, void* d_ws, size_t ws_size,
                              hipStream_t stream) {
  const float* query = (const float*)d_in[0];
  const float* facts = (const float*)d_in[1];
  const unsigned char* mask = (const unsigned char*)d_in[2];  // jnp bool_ = 1 byte
  const float* W1 = (const float*)d_in[3];
  const float* b1 = (const float*)d_in[4];
  const float* a1 = (const float*)d_in[5];
  const float* W2 = (const float*)d_in[6];
  const float* b2 = (const float*)d_in[7];
  const float* a2 = (const float*)d_in[8];
  const float* W3 = (const float*)d_in[9];
  const float* b3 = (const float*)d_in[10];
  float* out = (float*)d_out;
  (void)in_sizes; (void)n_in; (void)out_size; (void)ws_size;

  char* ws = (char*)d_ws;
  size_t off = 0;
  auto take = [&](size_t bytes) -> char* {
    char* p = ws + off;
    off += (bytes + 255) & ~(size_t)255;
    return p;
  };
  _Float16* h1  = (_Float16*)take((size_t)kRows * kL1 * sizeof(_Float16)); // 31.3 MiB
  _Float16* h2  = (_Float16*)take((size_t)kRows * kL2 * sizeof(_Float16)); // 15.6 MiB
  float* mean1  = (float*)take((size_t)kB * kL1 * sizeof(float));
  float* rstd1  = (float*)take((size_t)kB * kL1 * sizeof(float));
  float* mean2  = (float*)take((size_t)kB * kL2 * sizeof(float));
  float* rstd2  = (float*)take((size_t)kB * kL2 * sizeof(float));
  float* logits = (float*)take((size_t)kRows * sizeof(float));
  _Float16* w1t = (_Float16*)take((size_t)kL1 * kF * sizeof(_Float16));   // 80 KiB
  _Float16* w2t = (_Float16*)take((size_t)48 * 96 * sizeof(_Float16));    // 9 KiB

  attn_prep_weights<<<(kL1 * kF + 48 * 96 + 255) / 256, 256, 0, stream>>>(
      W1, W2, w1t, w2t);
  attn_gemm1<<<kRows / 32, 32, 0, stream>>>(query, facts, w1t, b1, h1);
  attn_stats<<<(kB * kL1) / 256, 256, 0, stream>>>(h1, mean1, rstd1, kL1);
  attn_gemm2<<<kRows / 32, 32, 0, stream>>>(h1, mean1, rstd1, a1, w2t, b2, h2);
  attn_stats<<<(kB * kL2) / 256, 256, 0, stream>>>(h2, mean2, rstd2, kL2);
  attn_logits<<<kRows / 256, 256, 0, stream>>>(h2, mean2, rstd2, a2, W3, b3,
                                               mask, logits);
  attn_softmax_scale<<<kB, 256, 0, stream>>>(logits, facts, out);
}
